// SRModel_7868380086839
// MI455X (gfx1250) — matrix operand order repaired and checked by reading, not compiled, not run on hardware
//
#include <hip/hip_runtime.h>
#include <hip/hip_bf16.h>
#include <math.h>

// ---------------------------------------------------------------------------
// MI455X (gfx1250) fused implicit-kernel SR model.
// Heavy path: per-pixel MLP done as fused WMMA GEMMs entirely in LDS.
// Activations use hardware transcendentals (v_exp_f32/v_sin_f32) — branch-free.
// ---------------------------------------------------------------------------

typedef __attribute__((ext_vector_type(8)))  _Float16 v8h;
typedef __attribute__((ext_vector_type(16))) _Float16 v16h;
typedef __attribute__((ext_vector_type(8)))  float    v8f;

#define HLs   80
#define WLs   80
#define HHs   320
#define WHs   320
#define NPIX  (HHs*WHs)   // 102400
#define LPIX  (HLs*WLs)   // 6400
#define FEATC 64
#define HID   256
#define KPAD0 128         // 106 padded to 128
#define KW_N  196
#define KW_NP 208         // 196 padded to 13 n-tiles
#define MT    64          // pixel rows per workgroup

// gelu tanh-approx rewritten via sigmoid identity: 0.5x(1+tanh(a)) = x/(1+e^{-2a})
// -> single branch-free v_exp_f32 instead of libm tanhf's divergent range reduction.
__device__ __forceinline__ float gelu_f(float x) {
  float a = -1.5957691216057308f * (x + 0.044715f * x * x * x);
  return x / (1.0f + __expf(a));
}
__device__ __forceinline__ float clamp01(float x) { return fminf(fmaxf(x, 0.f), 1.f); }

// ---------------------------------------------------------------------------
// Weight pack: src f32 [K][N] row-major -> dst f16 [Np][Kp] (transposed, padded)
// ---------------------------------------------------------------------------
__global__ void pack_wT(const float* __restrict__ src, _Float16* __restrict__ dst,
                        int K, int N, int Kp, int Np) {
  int i = blockIdx.x * blockDim.x + threadIdx.x;
  if (i >= Kp * Np) return;
  int n = i / Kp, k = i % Kp;
  float v = (k < K && n < N) ? src[k * N + n] : 0.0f;
  dst[n * Kp + k] = (_Float16)v;
}

// ---------------------------------------------------------------------------
// Encoder conv0: 3->64, 3x3 SAME (zero pad), GELU.  out: f16 [pix][64]
// ---------------------------------------------------------------------------
__global__ void enc_conv0(const float* __restrict__ lr, const float* __restrict__ w,
                          const float* __restrict__ b, _Float16* __restrict__ out) {
  int i = blockIdx.x * blockDim.x + threadIdx.x;
  if (i >= LPIX * FEATC) return;
  int oc = i & 63, pix = i >> 6;
  int y = pix / WLs, x = pix % WLs;
  float acc = b[oc];
  for (int ic = 0; ic < 3; ++ic)
    for (int ky = 0; ky < 3; ++ky) {
      int yy = y + ky - 1; if ((unsigned)yy >= (unsigned)HLs) continue;
      for (int kx = 0; kx < 3; ++kx) {
        int xx = x + kx - 1; if ((unsigned)xx >= (unsigned)WLs) continue;
        acc += lr[(ic * HLs + yy) * WLs + xx] * w[((oc * 3 + ic) * 3 + ky) * 3 + kx];
      }
    }
  out[pix * FEATC + oc] = (_Float16)gelu_f(acc);
}

// ---------------------------------------------------------------------------
// Encoder hidden conv: 64->64, 3x3 SAME, GELU. f16 in/out [pix][64]
// ---------------------------------------------------------------------------
__global__ void enc_convh(const _Float16* __restrict__ fin, const float* __restrict__ w,
                          const float* __restrict__ b, _Float16* __restrict__ out) {
  int i = blockIdx.x * blockDim.x + threadIdx.x;
  if (i >= LPIX * FEATC) return;
  int oc = i & 63, pix = i >> 6;
  int y = pix / WLs, x = pix % WLs;
  float acc = b[oc];
  for (int ky = 0; ky < 3; ++ky) {
    int yy = y + ky - 1; if ((unsigned)yy >= (unsigned)HLs) continue;
    for (int kx = 0; kx < 3; ++kx) {
      int xx = x + kx - 1; if ((unsigned)xx >= (unsigned)WLs) continue;
      const _Float16* src = fin + (yy * WLs + xx) * FEATC;
      const float* wp = w + (oc * FEATC) * 9 + ky * 3 + kx;
      for (int ic = 0; ic < FEATC; ++ic)
        acc += (float)src[ic] * wp[ic * 9];
    }
  }
  out[pix * FEATC + oc] = (_Float16)gelu_f(acc);
}

// ---------------------------------------------------------------------------
// Bicubic (Keys a=-0.5) 4x upsample, half-pixel centers, clamped borders.
// ---------------------------------------------------------------------------
__device__ __forceinline__ float cubw(float t) {
  t = fabsf(t); float t2 = t * t, t3 = t2 * t;
  if (t <= 1.f) return 1.5f * t3 - 2.5f * t2 + 1.f;
  if (t < 2.f)  return -0.5f * t3 + 2.5f * t2 - 4.f * t + 2.f;
  return 0.f;
}
__global__ void upsample4(const float* __restrict__ lr, float* __restrict__ up) {
  int i = blockIdx.x * blockDim.x + threadIdx.x;
  if (i >= 3 * NPIX) return;
  int c = i / NPIX, p = i % NPIX;
  int y = p / WHs, x = p % WHs;
  float sy = (y + 0.5f) * 0.25f - 0.5f, sx = (x + 0.5f) * 0.25f - 0.5f;
  int iy = (int)floorf(sy), ix = (int)floorf(sx);
  float acc = 0.f;
  for (int a = -1; a <= 2; ++a) {
    float wy = cubw(sy - (float)(iy + a));
    int yy = min(max(iy + a, 0), HLs - 1);
    for (int bb = -1; bb <= 2; ++bb) {
      float wx = cubw(sx - (float)(ix + bb));
      int xx = min(max(ix + bb, 0), WLs - 1);
      acc += wy * wx * lr[(c * HLs + yy) * WLs + xx];
    }
  }
  up[i] = acc;
}

// ---------------------------------------------------------------------------
// WMMA fragment loaders (wave32, v_wmma_f32_16x16x32_f16 layouts)
// ---------------------------------------------------------------------------
__device__ __forceinline__ v16h load_fragA(const _Float16* buf, int row0, int k0, int lane) {
  // 16-bit A 16x32: lane m=row0+(lane&15); contiguous K [16h..16h+15]
  int m = row0 + (lane & 15);
  int kb = k0 + ((lane >> 4) << 4);
  v8h c0 = *(const v8h*)(buf + m * HID + kb);
  v8h c1 = *(const v8h*)(buf + m * HID + kb + 8);
  v16h a;
#pragma unroll
  for (int i = 0; i < 8; ++i) { a[i] = c0[i]; a[i + 8] = c1[i]; }
  return a;
}
__device__ __forceinline__ v16h load_fragB(const _Float16* W, int Kp, int n0, int k0, int lane) {
  // B 32x16: lane n=n0+(lane&15); lane-half selects contiguous K 0-15 / 16-31
  int n = n0 + (lane & 15);
  int koff = k0 + ((lane >> 4) << 4);
  return *(const v16h*)(W + n * Kp + koff);
}

// One GEMM layer: out[M,N] = act(in[M,Kp] * W^T + bias); in/out in LDS (stride HID)
__device__ __forceinline__ void gemm_layer(const _Float16* in, _Float16* out,
                                           const _Float16* W, const float* bias, int biasN,
                                           int kTiles, int nTiles, int outCol0, bool doGelu,
                                           int wave, int lane) {
  __builtin_prefetch(W, 0, 1);  // global_prefetch_b8: warm L2/WGP$ for weights
  const int row0 = wave * 16;
  const int Kp = kTiles * 32;
  for (int nt = 0; nt < nTiles; ++nt) {
    v8f acc = {};
    for (int kt = 0; kt < kTiles; ++kt) {
      v16h a = load_fragA(in, row0, kt * 32, lane);
      v16h b = load_fragB(W, Kp, nt * 16, kt * 32, lane);
      acc = __builtin_amdgcn_wmma_f32_16x16x32_f16(
          /*neg_a=*/false, a, /*neg_b=*/false, b,
          /*c_mod=*/(short)0, acc, /*reuse_a=*/false, /*reuse_b=*/false);
    }
    int n = nt * 16 + (lane & 15);
    float bv = (n < biasN) ? bias[n] : 0.0f;
    int mrow = row0 + ((lane >> 4) << 3);
#pragma unroll
    for (int r = 0; r < 8; ++r) {
      float v = acc[r] + bv;
      if (doGelu) v = gelu_f(v);
      out[(mrow + r) * HID + outCol0 + n] = (_Float16)v;
    }
  }
}

// ---------------------------------------------------------------------------
// Fused MLP + heads + softmax + 7x7 dynamic filtering.  128 threads = 4 waves.
// ---------------------------------------------------------------------------
__global__ __launch_bounds__(128) void mlp_fused(
    const _Float16* __restrict__ feat,   // [6400][64] encoder output
    const float* __restrict__ lr,        // [3][80][80]
    const _Float16* __restrict__ W0t,    // [256][128]
    const _Float16* __restrict__ Wht,    // [4][256][256]
    const _Float16* __restrict__ Kt,     // [208][256]
    const _Float16* __restrict__ Gt,     // [16][256]
    const float* __restrict__ b0, const float* __restrict__ bh,
    const float* __restrict__ kb, const float* __restrict__ gb,
    float* __restrict__ res0)            // [3][102400]
{
  __shared__ _Float16 bufA[MT * HID];   // 32 KB
  __shared__ _Float16 bufB[MT * HID];   // 32 KB
  const int tid  = threadIdx.x;
  const int wave = tid >> 5, lane = tid & 31;
  const int tile0 = blockIdx.x * MT;

  // ---- build X (64 rows x 128 padded cols) into bufA ----
  for (int idx = tid; idx < MT * KPAD0; idx += 128) {
    int r = idx >> 7, c = idx & 127;
    int p  = tile0 + r;
    int hy = p / WHs, hx = p % WHs;
    float v = 0.0f;
    if (c < FEATC) {
      float lx = (hx + 0.5f) * 0.25f - 0.5f, ly = (hy + 0.5f) * 0.25f - 0.5f;
      int bxi = min(max((int)floorf(lx), 0), WLs - 1);
      int byi = min(max((int)floorf(ly), 0), HLs - 1);
      v = (float)feat[(byi * WLs + bxi) * FEATC + c];
    } else if (c < 104) {
      int c2 = c - FEATC, band = c2 >> 2, comp = c2 & 3;
      float freq = exp2f((float)band) * 3.14159265358979f;
      float hxn = (hx + 0.5f) * (2.0f / WHs) - 1.0f;
      float hyn = (hy + 0.5f) * (2.0f / HHs) - 1.0f;
      float arg = (comp < 2 ? hxn : hyn) * freq;
      v = (comp & 1) ? __cosf(arg) : __sinf(arg);
    } else if (c == 104) {
      float lx = (hx + 0.5f) * 0.25f - 0.5f; v = lx - floorf(lx);
    } else if (c == 105) {
      float ly = (hy + 0.5f) * 0.25f - 0.5f; v = ly - floorf(ly);
    }
    bufA[r * HID + c] = (_Float16)v;
  }
  __syncthreads();

  // ---- 5 GEMM layers, ping-pong LDS; each wave owns its 16-row stripe ----
  gemm_layer(bufA, bufB, W0t, b0, HID, 4, 16, 0, true, wave, lane);               // 106->256
  gemm_layer(bufB, bufA, Wht + 0 * HID * HID, bh + 0 * HID, HID, 8, 16, 0, true, wave, lane);
  gemm_layer(bufA, bufB, Wht + 1 * HID * HID, bh + 1 * HID, HID, 8, 16, 0, true, wave, lane);
  gemm_layer(bufB, bufA, Wht + 2 * HID * HID, bh + 2 * HID, HID, 8, 16, 0, true, wave, lane);
  gemm_layer(bufA, bufB, Wht + 3 * HID * HID, bh + 3 * HID, HID, 8, 16, 0, true, wave, lane);
  // heads: kern logits -> bufA[:,0:208), gate logits -> bufA[:,208:224)
  gemm_layer(bufB, bufA, Kt, kb, KW_N, 8, 13, 0,   false, wave, lane);
  gemm_layer(bufB, bufA, Gt, gb, 4,    8, 1,  KW_NP, false, wave, lane);
  __syncthreads();

  // ---- per-pixel epilogue: softmaxes + gated 7x7 dynamic filter ----
  if (tid < MT) {
    int r = tid, p = tile0 + r;
    int hy = p / WHs, hx = p % WHs;
    float lx = (hx + 0.5f) * 0.25f - 0.5f, ly = (hy + 0.5f) * 0.25f - 0.5f;
    int bxi = min(max((int)floorf(lx), 0), WLs - 1);
    int byi = min(max((int)floorf(ly), 0), HLs - 1);
    const _Float16* row = bufA + r * HID;

    // gate softmax over 4 heads
    float g[4], gm = -1e30f, gs = 0.f;
#pragma unroll
    for (int h = 0; h < 4; ++h) { g[h] = (float)row[KW_NP + h]; gm = fmaxf(gm, g[h]); }
#pragma unroll
    for (int h = 0; h < 4; ++h) { g[h] = __expf(g[h] - gm); gs += g[h]; }
#pragma unroll
    for (int h = 0; h < 4; ++h) g[h] /= gs;

    // per-head softmax over 49 taps, folded with gate into one 49-tap filter
    float wk[49];
#pragma unroll
    for (int k = 0; k < 49; ++k) wk[k] = 0.f;
    for (int h = 0; h < 4; ++h) {
      const _Float16* lg = row + h * 49;
      float m = -1e30f;
#pragma unroll
      for (int k = 0; k < 49; ++k) m = fmaxf(m, (float)lg[k]);
      float s = 0.f;
#pragma unroll
      for (int k = 0; k < 49; ++k) s += __expf((float)lg[k] - m);
      float inv = g[h] / s;
#pragma unroll
      for (int k = 0; k < 49; ++k) wk[k] += __expf((float)lg[k] - m) * inv;
    }

    // clamped 7x7 RGB gather
    float r0 = 0.f, r1 = 0.f, r2 = 0.f;
    int k = 0;
    for (int oy = -3; oy <= 3; ++oy) {
      int ny = min(max(byi + oy, 0), HLs - 1);
      for (int ox = -3; ox <= 3; ++ox, ++k) {
        int nx = min(max(bxi + ox, 0), WLs - 1);
        int off = ny * WLs + nx;
        float wv = wk[k];
        r0 += wv * lr[off];
        r1 += wv * lr[LPIX + off];
        r2 += wv * lr[2 * LPIX + off];
      }
    }
    res0[p] = r0; res0[NPIX + p] = r1; res0[2 * NPIX + p] = r2;
  }
}

// ---------------------------------------------------------------------------
// Grouped refiner convs (groups=3) at 320x320
// ---------------------------------------------------------------------------
__global__ void ref_conv0(const float* __restrict__ up, const float* __restrict__ res0,
                          const float* __restrict__ w, const float* __restrict__ b,
                          _Float16* __restrict__ out) {
  int i = blockIdx.x * blockDim.x + threadIdx.x;
  if (i >= 48 * NPIX) return;
  int oc = i / NPIX, p = i % NPIX;
  int y = p / WHs, x = p % WHs;
  int grp = oc >> 4;
  float acc = b[oc];
  for (int icl = 0; icl < 2; ++icl) {
    int ic = grp * 2 + icl;
    const float* src = (ic < 3) ? (up + ic * NPIX) : (res0 + (ic - 3) * NPIX);
    const float* wp = w + (oc * 2 + icl) * 9;
    for (int ky = 0; ky < 3; ++ky) {
      int yy = y + ky - 1; if ((unsigned)yy >= (unsigned)HHs) continue;
      for (int kx = 0; kx < 3; ++kx) {
        int xx = x + kx - 1; if ((unsigned)xx >= (unsigned)WHs) continue;
        acc += src[yy * WHs + xx] * wp[ky * 3 + kx];
      }
    }
  }
  out[i] = (_Float16)gelu_f(acc);
}

__global__ void ref_conv1(const _Float16* __restrict__ hin, const float* __restrict__ w,
                          const float* __restrict__ b, _Float16* __restrict__ out) {
  int i = blockIdx.x * blockDim.x + threadIdx.x;
  if (i >= 48 * NPIX) return;
  int oc = i / NPIX, p = i % NPIX;
  int y = p / WHs, x = p % WHs;
  int grp = oc >> 4;
  float acc = b[oc];
  for (int ky = 0; ky < 3; ++ky) {
    int yy = y + ky - 1; if ((unsigned)yy >= (unsigned)HHs) continue;
    for (int kx = 0; kx < 3; ++kx) {
      int xx = x + kx - 1; if ((unsigned)xx >= (unsigned)WHs) continue;
      for (int icl = 0; icl < 16; ++icl) {
        int ic = grp * 16 + icl;
        acc += (float)hin[ic * NPIX + yy * WHs + xx] * w[((oc * 16 + icl) * 3 + ky) * 3 + kx];
      }
    }
  }
  out[i] = (_Float16)gelu_f(acc);
}

__global__ void ref_conv2(const _Float16* __restrict__ hin, const float* __restrict__ w,
                          const float* __restrict__ b, float* __restrict__ out) {
  int i = blockIdx.x * blockDim.x + threadIdx.x;
  if (i >= 3 * NPIX) return;
  int oc = i / NPIX, p = i % NPIX;
  int y = p / WHs, x = p % WHs;
  float acc = b[oc];
  for (int ky = 0; ky < 3; ++ky) {
    int yy = y + ky - 1; if ((unsigned)yy >= (unsigned)HHs) continue;
    for (int kx = 0; kx < 3; ++kx) {
      int xx = x + kx - 1; if ((unsigned)xx >= (unsigned)WHs) continue;
      for (int icl = 0; icl < 16; ++icl) {
        int ic = oc * 16 + icl;
        acc += (float)hin[ic * NPIX + yy * WHs + xx] * w[((oc * 16 + icl) * 3 + ky) * 3 + kx];
      }
    }
  }
  out[i] = acc;
}

// ---------------------------------------------------------------------------
// Final fusion: clip + color-preserving gray-delta
// ---------------------------------------------------------------------------
__global__ void fuse_out(const float* __restrict__ up, const float* __restrict__ res0,
                         const float* __restrict__ rref, float* __restrict__ out) {
  int p = blockIdx.x * blockDim.x + threadIdx.x;
  if (p >= NPIX) return;
  float u0 = up[p], u1 = up[NPIX + p], u2 = up[2 * NPIX + p];
  float s0 = clamp01(u0 + res0[p]            + rref[p]);
  float s1 = clamp01(u1 + res0[NPIX + p]     + rref[NPIX + p]);
  float s2 = clamp01(u2 + res0[2 * NPIX + p] + rref[2 * NPIX + p]);
  float d = (0.2989f * s0 + 0.587f * s1 + 0.114f * s2)
          - (0.2989f * u0 + 0.587f * u1 + 0.114f * u2);
  out[p]            = clamp01(u0 + d);
  out[NPIX + p]     = clamp01(u1 + d);
  out[2 * NPIX + p] = clamp01(u2 + d);
}

// ---------------------------------------------------------------------------
extern "C" void kernel_launch(void* const* d_in, const int* in_sizes, int n_in,
                              void* d_out, int out_size, void* d_ws, size_t ws_size,
                              hipStream_t stream) {
  const float* lr     = (const float*)d_in[0];
  const float* enc_w0 = (const float*)d_in[1];
  const float* enc_b0 = (const float*)d_in[2];
  const float* enc_wh = (const float*)d_in[3];
  const float* enc_bh = (const float*)d_in[4];
  const float* mlp_w0 = (const float*)d_in[5];
  const float* mlp_b0 = (const float*)d_in[6];
  const float* mlp_wh = (const float*)d_in[7];
  const float* mlp_bh = (const float*)d_in[8];
  const float* kern_w = (const float*)d_in[9];
  const float* kern_b = (const float*)d_in[10];
  const float* gate_w = (const float*)d_in[11];
  const float* gate_b = (const float*)d_in[12];
  const float* ref_w0 = (const float*)d_in[13];
  const float* ref_b0 = (const float*)d_in[14];
  const float* ref_w1 = (const float*)d_in[15];
  const float* ref_b1 = (const float*)d_in[16];
  const float* ref_w2 = (const float*)d_in[17];
  const float* ref_b2 = (const float*)d_in[18];
  float* out = (float*)d_out;

  // ---- workspace layout ----
  char* ws = (char*)d_ws;
  size_t off = 0;
  auto alloc = [&](size_t bytes) { char* p = ws + off; off += (bytes + 511) & ~size_t(511); return p; };
  _Float16* W0t    = (_Float16*)alloc(HID * KPAD0 * 2);        // [256][128]
  _Float16* Wht    = (_Float16*)alloc(4 * HID * HID * 2);      // [4][256][256]
  _Float16* Kt     = (_Float16*)alloc(KW_NP * HID * 2);        // [208][256]
  _Float16* Gt     = (_Float16*)alloc(16 * HID * 2);           // [16][256]
  _Float16* feat_a = (_Float16*)alloc(LPIX * FEATC * 2);
  _Float16* feat_b = (_Float16*)alloc(LPIX * FEATC * 2);
  float*    up     = (float*)alloc(3 * NPIX * 4);
  float*    res0   = (float*)alloc(3 * NPIX * 4);
  _Float16* hr1    = (_Float16*)alloc(48 * NPIX * 2);
  _Float16* hr2    = (_Float16*)alloc(48 * NPIX * 2);
  float*    rref   = (float*)alloc(3 * NPIX * 4);
  (void)ws_size; (void)in_sizes; (void)n_in; (void)out_size;

  // ---- pack / transpose weights to f16 [N][Kp] ----
  pack_wT<<<(HID * KPAD0 + 255) / 256, 256, 0, stream>>>(mlp_w0, W0t, 106, HID, KPAD0, HID);
  for (int l = 0; l < 4; ++l)
    pack_wT<<<(HID * HID + 255) / 256, 256, 0, stream>>>(
        mlp_wh + l * HID * HID, Wht + l * HID * HID, HID, HID, HID, HID);
  pack_wT<<<(KW_NP * HID + 255) / 256, 256, 0, stream>>>(kern_w, Kt, HID, KW_N, HID, KW_NP);
  pack_wT<<<(16 * HID + 255) / 256, 256, 0, stream>>>(gate_w, Gt, HID, 4, HID, 16);

  // ---- encoder ----
  enc_conv0<<<(LPIX * FEATC + 255) / 256, 256, 0, stream>>>(lr, enc_w0, enc_b0, feat_a);
  enc_convh<<<(LPIX * FEATC + 255) / 256, 256, 0, stream>>>(feat_a, enc_wh + 0 * FEATC * FEATC * 9, enc_bh + 0 * FEATC, feat_b);
  enc_convh<<<(LPIX * FEATC + 255) / 256, 256, 0, stream>>>(feat_b, enc_wh + 1 * FEATC * FEATC * 9, enc_bh + 1 * FEATC, feat_a);
  enc_convh<<<(LPIX * FEATC + 255) / 256, 256, 0, stream>>>(feat_a, enc_wh + 2 * FEATC * FEATC * 9, enc_bh + 2 * FEATC, feat_b);

  // ---- bicubic upsample ----
  upsample4<<<(3 * NPIX + 255) / 256, 256, 0, stream>>>(lr, up);

  // ---- fused WMMA MLP + dynamic filtering ----
  mlp_fused<<<NPIX / MT, 128, 0, stream>>>(feat_b, lr, W0t, Wht, Kt, Gt,
                                           mlp_b0, mlp_bh, kern_b, gate_b, res0);

  // ---- refiner ----
  ref_conv0<<<(48 * NPIX + 255) / 256, 256, 0, stream>>>(up, res0, ref_w0, ref_b0, hr1);
  ref_conv1<<<(48 * NPIX + 255) / 256, 256, 0, stream>>>(hr1, ref_w1, ref_b1, hr2);
  ref_conv2<<<(3 * NPIX + 255) / 256, 256, 0, stream>>>(hr2, ref_w2, ref_b2, rref);

  // ---- fusion ----
  fuse_out<<<(NPIX + 255) / 256, 256, 0, stream>>>(up, res0, rref, out);
}
